// MPNN_2_38792144617822
// MI455X (gfx1250) — compile-verified, matrix operand order checked by reading, not run on hardware
//
#include <hip/hip_runtime.h>

typedef __attribute__((ext_vector_type(16))) _Float16 v16h;
typedef __attribute__((ext_vector_type(8)))  float    v8f;

#define FEAT 128
#define AS   144   // LDS row stride in halves (288B: 32B-aligned, spreads banks)

__device__ __forceinline__ unsigned fkey(float f) {
    unsigned u = __float_as_uint(f);
    return (u & 0x80000000u) ? ~u : (u | 0x80000000u);
}
__device__ __forceinline__ float funkey(unsigned k) {
    return __uint_as_float((k & 0x80000000u) ? (k & 0x7fffffffu) : ~k);
}
__device__ __forceinline__ float sigmoidf_(float x) {
    return 1.0f / (1.0f + __expf(-x));
}

// ---------------- small prep kernels ----------------

__global__ void k_zero(float* agg, float* denom, unsigned* key, int n_nodes) {
    int i = blockIdx.x * blockDim.x + threadIdx.x;
    int stride = gridDim.x * blockDim.x;
    int total = n_nodes * FEAT;
    for (int j = i; j < total; j += stride) agg[j] = 0.0f;
    for (int j = i; j < n_nodes; j += stride) { denom[j] = 0.0f; key[j] = 0u; }
}

// wT[j][k] = (f16) w[k][j], first 128 rows of w (row stride FEAT)
__global__ void k_wconv(const float* __restrict__ w, _Float16* __restrict__ wT) {
    int i = blockIdx.x * blockDim.x + threadIdx.x;
    if (i < FEAT * FEAT) {
        int j = i >> 7, k = i & 127;
        wT[i] = (_Float16)w[k * FEAT + j];
    }
}

// pre[r][j] = sum_k etab[r][k] * w1[(128+k)][j] + b1[j]
__global__ void k_pre_e(const float* __restrict__ etab, const float* __restrict__ w1,
                        const float* __restrict__ b1, float* __restrict__ pre,
                        int nrel, int edim) {
    int i = blockIdx.x * blockDim.x + threadIdx.x;
    if (i >= nrel * FEAT) return;
    int r = i >> 7, j = i & 127;
    float acc = b1[j];
    for (int k = 0; k < edim; ++k)
        acc += etab[r * edim + k] * w1[(FEAT + k) * FEAT + j];
    pre[i] = acc;
}

// dd[n] = h[n] . w3[128:256] + b3   (one wave32 per node)
__global__ void __launch_bounds__(256)
k_dstdot(const float* __restrict__ h, const float* __restrict__ w3,
         const float* __restrict__ b3, float* __restrict__ dd, int n_nodes) {
    int wid  = (blockIdx.x * blockDim.x + threadIdx.x) >> 5;
    int lane = threadIdx.x & 31;
    if (wid >= n_nodes) return;
    float4 hv = *(const float4*)(h + (size_t)wid * FEAT + lane * 4);
    float4 wv = *(const float4*)(w3 + FEAT + lane * 4);
    float p = hv.x * wv.x + hv.y * wv.y + hv.z * wv.z + hv.w * wv.w;
    for (int off = 16; off >= 1; off >>= 1) p += __shfl_xor(p, off, 32);
    if (lane == 0) dd[wid] = p + b3[0];
}

// ---------------- edge kernel: gather + msg GEMM (WMMA) + logit + segmax ----

__global__ void __launch_bounds__(128)
k_edge_msg(const float* __restrict__ h, const int* __restrict__ src,
           const int* __restrict__ dst, const int* __restrict__ et,
           const _Float16* __restrict__ wT1, const float* __restrict__ pre_e,
           const float* __restrict__ w3, const float* __restrict__ dd,
           _Float16* __restrict__ msg, float* __restrict__ logit,
           unsigned* __restrict__ segkey, const int* __restrict__ nwp,
           int add_src, int add_dst, int nE) {
    __shared__ _Float16 Ah[16 * AS];
    __shared__ float logit_s[16];
    __shared__ int et_s[16], dstg_s[16], srcg_s[16];

    int tid = threadIdx.x;
    int ebase = blockIdx.x * 16;
    int nw = *nwp;

    if (tid < 16) {
        int e = ebase + tid;
        int ec = e < nE ? e : nE - 1;
        et_s[tid]   = et[ec];
        dstg_s[tid] = dst[ec] + (add_dst ? nw : 0);
        logit_s[tid] = 0.0f;
    } else if (tid < 32) {
        int r = tid - 16;
        int e = ebase + r;
        int ec = e < nE ? e : nE - 1;
        srcg_s[r] = src[ec] + (add_src ? nw : 0);
    }
    __syncthreads();

    // stage gathered rows as f16 into LDS: 8 threads per row, 16 cols each
    {
        int r = tid >> 3, g = tid & 7;
        const float* row = h + (size_t)srcg_s[r] * FEAT + g * 16;
        _Float16* o = &Ah[r * AS + g * 16];
#pragma unroll
        for (int q = 0; q < 4; ++q) {
            float4 v = *(const float4*)(row + q * 4);
            o[q * 4 + 0] = (_Float16)v.x; o[q * 4 + 1] = (_Float16)v.y;
            o[q * 4 + 2] = (_Float16)v.z; o[q * 4 + 3] = (_Float16)v.w;
        }
    }
    __syncthreads();

    int wave = tid >> 5, lane = tid & 31;
    int half = lane >> 4, l16 = lane & 15;
    int n0 = wave * 32;
    v8f c0 = {}, c1 = {};
    const _Float16* arow = &Ah[l16 * AS + half * 16];
#pragma unroll
    for (int kk = 0; kk < FEAT; kk += 32) {
        v16h a   = *(const v16h*)(arow + kk);
        v16h b0v = *(const v16h*)(wT1 + (size_t)(n0 + l16) * FEAT + half * 16 + kk);
        v16h b1v = *(const v16h*)(wT1 + (size_t)(n0 + 16 + l16) * FEAT + half * 16 + kk);
        c0 = __builtin_amdgcn_wmma_f32_16x16x32_f16(false, a, false, b0v, (short)0, c0, false, false);
        c1 = __builtin_amdgcn_wmma_f32_16x16x32_f16(false, a, false, b1v, (short)0, c1, false, false);
    }

    int colA = n0 + l16, colB = n0 + 16 + l16;
    float w3a0 = w3[colA], w3a1 = w3[colB];
#pragma unroll
    for (int i = 0; i < 8; ++i) {
        int m = i + 8 * half;
        int e = ebase + m;
        int etv = et_s[m];
        float v0 = sigmoidf_(c0[i] + pre_e[etv * FEAT + colA]);
        float v1 = sigmoidf_(c1[i] + pre_e[etv * FEAT + colB]);
        if (e < nE) {
            msg[(size_t)e * FEAT + colA] = (_Float16)v0;
            msg[(size_t)e * FEAT + colB] = (_Float16)v1;
        }
        float p = v0 * w3a0 + v1 * w3a1;              // partial msg . w3a
        for (int off = 8; off >= 1; off >>= 1) p += __shfl_xor(p, off, 32);
        if (l16 == 0) atomicAdd(&logit_s[m], p);      // lanes 0 and 16: rows i, i+8
    }
    __syncthreads();

    if (tid < 16) {
        int e = ebase + tid;
        if (e < nE) {
            float lg = logit_s[tid] + dd[dstg_s[tid]];
            logit[e] = lg;
            atomicMax(segkey + dstg_s[tid], fkey(lg));
        }
    }
}

// ---------------- edge pass 2: exp + atomic aggregation ----------------

__global__ void __launch_bounds__(256)
k_edge_agg(const _Float16* __restrict__ msg, const float* __restrict__ logit,
           const unsigned* __restrict__ segkey, const int* __restrict__ dst,
           float* __restrict__ denom, float* __restrict__ agg,
           const int* __restrict__ nwp, int add_dst, int nE) {
    int wid  = (blockIdx.x * blockDim.x + threadIdx.x) >> 5;
    int lane = threadIdx.x & 31;
    if (wid >= nE) return;
    int d = dst[wid] + (add_dst ? *nwp : 0);
    float ex = 0.0f;
    if (lane == 0) ex = __expf(logit[wid] - funkey(segkey[d]));
    ex = __shfl(ex, 0, 32);
    if (lane == 0) atomicAdd(denom + d, ex);
    const _Float16* mrow = msg + (size_t)wid * FEAT + lane * 4;
    float* arow = agg + (size_t)d * FEAT + lane * 4;
#pragma unroll
    for (int q = 0; q < 4; ++q) atomicAdd(arow + q, ex * (float)mrow[q]);
}

// ---------------- node kernel: normalize + node GEMM (WMMA) + predicated out

__global__ void __launch_bounds__(128)
k_node(const float* __restrict__ h_in, const float* __restrict__ agg,
       const float* __restrict__ denom, const _Float16* __restrict__ wT2,
       const float* __restrict__ b2, float* __restrict__ h_out,
       float cm, float cs, int n_nodes) {
    __shared__ _Float16 Ah[16 * AS];
    __shared__ float flag_s[16];
    int tid = threadIdx.x;
    int nbase = blockIdx.x * 16;
    {
        int r = tid >> 3, g = tid & 7;
        int nn = nbase + r; if (nn >= n_nodes) nn = n_nodes - 1;
        float dgn = denom[nn];
        if (g == 0) flag_s[r] = dgn;
        float inv = dgn > 0.0f ? cm / dgn : 0.0f;
        const float* ar = agg  + (size_t)nn * FEAT + g * 16;
        const float* hr = h_in + (size_t)nn * FEAT + g * 16;
        _Float16* o = &Ah[r * AS + g * 16];
#pragma unroll
        for (int q = 0; q < 4; ++q) {
            float4 av = *(const float4*)(ar + q * 4);
            float4 hv = *(const float4*)(hr + q * 4);
            o[q * 4 + 0] = (_Float16)(av.x * inv + cs * hv.x);
            o[q * 4 + 1] = (_Float16)(av.y * inv + cs * hv.y);
            o[q * 4 + 2] = (_Float16)(av.z * inv + cs * hv.z);
            o[q * 4 + 3] = (_Float16)(av.w * inv + cs * hv.w);
        }
    }
    __syncthreads();

    int wave = tid >> 5, lane = tid & 31;
    int half = lane >> 4, l16 = lane & 15;
    int n0 = wave * 32;
    v8f c0 = {}, c1 = {};
    const _Float16* arow = &Ah[l16 * AS + half * 16];
#pragma unroll
    for (int kk = 0; kk < FEAT; kk += 32) {
        v16h a   = *(const v16h*)(arow + kk);
        v16h b0v = *(const v16h*)(wT2 + (size_t)(n0 + l16) * FEAT + half * 16 + kk);
        v16h b1v = *(const v16h*)(wT2 + (size_t)(n0 + 16 + l16) * FEAT + half * 16 + kk);
        c0 = __builtin_amdgcn_wmma_f32_16x16x32_f16(false, a, false, b0v, (short)0, c0, false, false);
        c1 = __builtin_amdgcn_wmma_f32_16x16x32_f16(false, a, false, b1v, (short)0, c1, false, false);
    }

    int colA = n0 + l16, colB = n0 + 16 + l16;
    float bA = b2[colA], bB = b2[colB];
#pragma unroll
    for (int i = 0; i < 8; ++i) {
        int m = i + 8 * half;
        int n = nbase + m;
        if (n >= n_nodes) continue;
        bool upd = flag_s[m] > 0.0f;
        float vA = sigmoidf_(c0[i] + bA);
        float vB = sigmoidf_(c1[i] + bB);
        size_t base = (size_t)n * FEAT;
        h_out[base + colA] = upd ? vA : h_in[base + colA];
        h_out[base + colB] = upd ? vB : h_in[base + colB];
    }
}

// ---------------- host ----------------

extern "C" void kernel_launch(void* const* d_in, const int* in_sizes, int n_in,
                              void* d_out, int out_size, void* d_ws, size_t ws_size,
                              hipStream_t stream) {
    const float* features = (const float*)d_in[0];
    const int*   e1_src   = (const int*)d_in[1];
    const int*   e1_dst   = (const int*)d_in[2];
    const int*   e1_type  = (const int*)d_in[3];
    const int*   e2_src   = (const int*)d_in[4];
    const int*   e2_dst   = (const int*)d_in[5];
    const int*   e2_type  = (const int*)d_in[6];
    const int*   nwp      = (const int*)d_in[7];
    const float* w1_wkr = (const float*)d_in[9];
    const float* b1_wkr = (const float*)d_in[10];
    const float* w1_tsk = (const float*)d_in[11];
    const float* b1_tsk = (const float*)d_in[12];
    const float* w2_wkr = (const float*)d_in[13];
    const float* b2_wkr = (const float*)d_in[14];
    const float* w2_tsk = (const float*)d_in[15];
    const float* b2_tsk = (const float*)d_in[16];
    const float* w3_wkr = (const float*)d_in[17];
    const float* b3_wkr = (const float*)d_in[18];
    const float* w3_tsk = (const float*)d_in[19];
    const float* b3_tsk = (const float*)d_in[20];
    const float* e_wkr  = (const float*)d_in[21];
    const float* e_tsk  = (const float*)d_in[22];

    const int N  = in_sizes[0] / FEAT;
    const int E1 = in_sizes[1];
    const int E2 = in_sizes[4];
    const int edim = in_sizes[9] / FEAT - FEAT;   // (FEAT+EDIM)*FEAT
    const int nrel = in_sizes[21] / edim;
    const int Emax = E1 > E2 ? E1 : E2;

    // workspace carve (256B aligned)
    char* p = (char*)d_ws;
    auto take = [&](size_t bytes) { void* r = (void*)p; p += (bytes + 255) & ~(size_t)255; return r; };
    _Float16* msg    = (_Float16*)take((size_t)Emax * FEAT * 2);
    float*    logit  = (float*)   take((size_t)Emax * 4);
    unsigned* segkey = (unsigned*)take((size_t)N * 4);
    float*    denom  = (float*)   take((size_t)N * 4);
    float*    agg    = (float*)   take((size_t)N * FEAT * 4);
    float*    dd     = (float*)   take((size_t)N * 4);
    _Float16* wT1    = (_Float16*)take((size_t)FEAT * FEAT * 2);
    _Float16* wT2    = (_Float16*)take((size_t)FEAT * FEAT * 2);
    float*    pre    = (float*)   take((size_t)nrel * FEAT * 4);
    float*    h1     = (float*)   take((size_t)N * FEAT * 4);

    auto run_phase = [&](const float* h_in, float* h_out,
                         const int* src, const int* dst, const int* et, int E,
                         const float* w1, const float* b1, const float* w2,
                         const float* b2, const float* w3, const float* b3,
                         const float* etab, int add_src, int add_dst,
                         float cm, float cs) {
        int zgrid = (N * FEAT + 255) / 256;
        k_zero<<<zgrid, 256, 0, stream>>>(agg, denom, segkey, N);
        k_wconv<<<(FEAT * FEAT + 255) / 256, 256, 0, stream>>>(w1, wT1);
        k_wconv<<<(FEAT * FEAT + 255) / 256, 256, 0, stream>>>(w2, wT2);
        k_pre_e<<<(nrel * FEAT + 255) / 256, 256, 0, stream>>>(etab, w1, b1, pre, nrel, edim);
        k_dstdot<<<(N + 7) / 8, 256, 0, stream>>>(h_in, w3, b3, dd, N);
        k_edge_msg<<<(E + 15) / 16, 128, 0, stream>>>(h_in, src, dst, et, wT1, pre,
                                                      w3, dd, msg, logit, segkey,
                                                      nwp, add_src, add_dst, E);
        k_edge_agg<<<(E + 7) / 8, 256, 0, stream>>>(msg, logit, segkey, dst,
                                                    denom, agg, nwp, add_dst, E);
        k_node<<<(N + 15) / 16, 128, 0, stream>>>(h_in, agg, denom, wT2, b2, h_out,
                                                  cm, cs, N);
    };

    // phase 1: tasks -> workers  (src += NW, dst as-is), c_msg=1.0, c_self=0.5
    run_phase(features, h1, e1_src, e1_dst, e1_type, E1,
              w1_wkr, b1_wkr, w2_wkr, b2_wkr, w3_wkr, b3_wkr, e_wkr,
              1, 0, 1.0f, 0.5f);
    // phase 2: workers -> tasks  (dst += NW), c_msg=0.7, c_self=0.3
    run_phase(h1, (float*)d_out, e2_src, e2_dst, e2_type, E2,
              w1_tsk, b1_tsk, w2_tsk, b2_tsk, w3_tsk, b3_tsk, e_tsk,
              0, 1, 0.7f, 0.3f);
}
